// ContinuousConv_57578331570481
// MI455X (gfx1250) — compile-verified, matrix operand order checked by reading, not run on hardware
//
#include <hip/hip_runtime.h>
#include <hip/hip_bf16.h>
#include <math.h>

typedef __attribute__((ext_vector_type(16))) _Float16 v16h;
typedef __attribute__((ext_vector_type(8)))  float    v8f;

#define N_NODES   20000
#define N_EDGES   220000
#define CIN       16
#define COUT      32
#define NTILES    ((N_EDGES + 15) / 16)
#define MAIN_BLOCKS 256
#define WPB       8      // waves per block (wave32)

// ---------------------------------------------------------------------------
// init: zero accumulator output and per-row edge counts (in workspace)
// ---------------------------------------------------------------------------
__global__ __launch_bounds__(256) void cc_init(float* __restrict__ out,
                                               float* __restrict__ cnt) {
    int i = blockIdx.x * 256 + threadIdx.x;
    if (i < N_NODES * COUT) out[i] = 0.0f;
    int j = i - N_NODES * COUT;
    if (j >= 0 && j < N_NODES) cnt[j] = 0.0f;
}

// ---------------------------------------------------------------------------
// main edge kernel: per 16-edge tile, D(16x32) += A(16x1024) x B(1024x32)
// A[e, cell*16+ci] = g[e,cell] * f[col[e],ci], built in registers from the
// separable trilinear weights g = X[kx]*Y[ky]*Z[kz] (window folded into X).
// B = all 64 cell filters, f16, pre-swizzled into the WMMA B-fragment layout
// and cached in LDS (64 KB). 2 cells per v_wmma_f32_16x16x32_f16 (K=32).
// The (kx,ky) loop is fully unrolled (all weight indices compile-time -> no
// selects/branches); per-body asm scheduling fences stop the scheduler from
// hoisting all B-fragment LDS loads (the round-1 spill cause).
// ---------------------------------------------------------------------------
__global__ __launch_bounds__(256) void cc_edges(const float* __restrict__ pos,
                                                const float* __restrict__ feat,
                                                const float* __restrict__ filt,
                                                const int*   __restrict__ rowp,
                                                const int*   __restrict__ colp,
                                                float* __restrict__ out,
                                                float* __restrict__ cnt) {
    // [s=32 k-steps][t=2 n-tiles][lane=32][16 halfs]  == 64 KB
    __shared__ _Float16 ldsB[32 * 2 * 32 * 16];

    const int tid = threadIdx.x;

    // ---- cooperative build of B fragments (f32 -> f16, swizzled, packed) --
    {
        unsigned int* ldsB32 = (unsigned int*)ldsB;
        for (int d = tid; d < 32 * 2 * 32 * 16 / 2; d += 256) {
            int h2   = d & 7;                 // dword within 16-half fragment
            int lane = (d >> 3) & 31;
            int t    = (d >> 8) & 1;
            int s    = d >> 9;
            int h    = 2 * h2;                // first half index (even)
            int slot = (h >> 3) & 1;          // which of the 2 packed cells
            int cell = 2 * s + slot;          // grid cell 0..63
            int ci   = (h & 7) + ((lane >> 4) << 3);
            int n    = 16 * t + (lane & 15);
            _Float16 a0 = (_Float16)filt[(cell * CIN + ci) * COUT + n];
            _Float16 a1 = (_Float16)filt[(cell * CIN + ci + 1) * COUT + n];
            unsigned int p0 = __builtin_bit_cast(unsigned short, a0);
            unsigned int p1 = __builtin_bit_cast(unsigned short, a1);
            ldsB32[d] = (p1 << 16) | p0;
        }
    }
    __syncthreads();

    const int   lane = tid & 31;
    const int   wave = blockIdx.x * WPB + (tid >> 5);
    const int   wstride = gridDim.x * WPB;
    const v16h* bl = ((const v16h*)ldsB) + lane;   // per-lane fragment base

    for (int tile = wave; tile < NTILES; tile += wstride) {
        // ---- per-lane edge geometry (lanes l and l+16 share edge l&15) ----
        const int  m    = lane & 15;
        const int  e    = tile * 16 + m;
        const bool live = (e < N_EDGES);
        const int  eu   = live ? e : (N_EDGES - 1);
        const int  ri   = rowp[eu];
        const int  cj   = colp[eu];

        float rx = pos[3 * cj + 0] - pos[3 * ri + 0];
        float ry = pos[3 * cj + 1] - pos[3 * ri + 1];
        float rz = pos[3 * cj + 2] - pos[3 * ri + 2];
        float d2 = rx * rx + ry * ry + rz * rz;

        float w1     = 1.0f - d2 * 4.0f;                    // 1 - d2/R^2
        float window = (live && (d2 < 0.25f)) ? (w1 * w1 * w1) : 0.0f;
        float nrm    = sqrtf(d2);
        float sc     = tanhf(nrm) / (nrm + 1e-8f);
        float gx = (rx * sc + 1.0f) * 1.5f;                 // (D-1)/2 = 1.5
        float gy = (ry * sc + 1.0f) * 1.5f;
        float gz = (rz * sc + 1.0f) * 1.5f;

        // separable trilinear weights; window folded into X. All indices
        // below are compile-time, so these live purely in registers.
        float X[4], Y[4], Z[4];
        {
            float x0f = floorf(gx); float xd = gx - x0f;
            float y0f = floorf(gy); float yd = gy - y0f;
            float z0f = floorf(gz); float zd = gz - z0f;
            int x0 = (int)x0f; x0 = x0 < 0 ? 0 : (x0 > 3 ? 3 : x0);
            int y0 = (int)y0f; y0 = y0 < 0 ? 0 : (y0 > 3 ? 3 : y0);
            int z0 = (int)z0f; z0 = z0 < 0 ? 0 : (z0 > 3 ? 3 : z0);
            int x1 = x0 + 1 > 3 ? 3 : x0 + 1;
            int y1 = y0 + 1 > 3 ? 3 : y0 + 1;
            int z1 = z0 + 1 > 3 ? 3 : z0 + 1;
#pragma unroll
            for (int j = 0; j < 4; ++j) {
                X[j] = (((j == x0) ? (1.0f - xd) : 0.0f) +
                        ((j == x1) ? xd : 0.0f)) * window;
                Y[j] = ((j == y0) ? (1.0f - yd) : 0.0f) +
                       ((j == y1) ? yd : 0.0f);
                Z[j] = ((j == z0) ? (1.0f - zd) : 0.0f) +
                       ((j == z1) ? zd : 0.0f);
            }
        }

        // features for this lane's CIN half, converted to f16 once
        const int ci0 = (lane >> 4) << 3;
        _Float16 fh[8];
#pragma unroll
        for (int h = 0; h < 8; ++h)
            fh[h] = (_Float16)(live ? feat[cj * CIN + ci0 + h] : 0.0f);

        // ---- WMMA main loop: 16 (kx,ky) x 2 z-pairs x 2 n-tiles = 64 wmma -
        v8f c0 = {}; v8f c1 = {};
#pragma unroll
        for (int p = 0; p < 16; ++p) {
            const int kx = p >> 2, ky = p & 3;       // compile-time constants
            float gxy = X[kx] * Y[ky];
            _Float16 g0 = (_Float16)(gxy * Z[0]);
            _Float16 g1 = (_Float16)(gxy * Z[1]);
            _Float16 g2 = (_Float16)(gxy * Z[2]);
            _Float16 g3 = (_Float16)(gxy * Z[3]);

            v16h a0, a1;                 // s = 2p (cells 4p,4p+1), s = 2p+1
#pragma unroll
            for (int h = 0; h < 8; ++h) {
                a0[h]     = g0 * fh[h];
                a0[h + 8] = g1 * fh[h];
                a1[h]     = g2 * fh[h];
                a1[h + 8] = g3 * fh[h];
            }
            // fragment element index: (s*2 + t)*32 + lane  ->  (4p+..)*32
            v16h b00 = bl[(4 * p + 0) * 32];
            v16h b01 = bl[(4 * p + 1) * 32];
            v16h b10 = bl[(4 * p + 2) * 32];
            v16h b11 = bl[(4 * p + 3) * 32];
            c0 = __builtin_amdgcn_wmma_f32_16x16x32_f16(false, a0, false, b00,
                                                        (short)0, c0, false, false);
            c1 = __builtin_amdgcn_wmma_f32_16x16x32_f16(false, a0, false, b01,
                                                        (short)0, c1, false, false);
            c0 = __builtin_amdgcn_wmma_f32_16x16x32_f16(false, a1, false, b10,
                                                        (short)0, c0, false, false);
            c1 = __builtin_amdgcn_wmma_f32_16x16x32_f16(false, a1, false, b11,
                                                        (short)0, c1, false, false);
            // scheduling fence: keep each body's LDS loads in its own body so
            // the scheduler cannot hoist all 64 fragment loads (reg spills)
            asm volatile("" ::: "memory");
        }

        // ---- scatter: D layout lane<16 -> M=j, lane>=16 -> M=8+j ----------
        const int ncol  = lane & 15;
        const int mbase = (lane < 16) ? 0 : 8;
#pragma unroll
        for (int j = 0; j < 8; ++j) {
            int ej = tile * 16 + mbase + j;
            if (ej < N_EDGES) {
                int rj = rowp[ej];
                atomicAdd(&out[rj * COUT + ncol], c0[j]);
                atomicAdd(&out[rj * COUT + 16 + ncol], c1[j]);
            }
        }
        if ((lane < 16) && live) atomicAdd(&cnt[ri], 1.0f);
    }
}

// ---------------------------------------------------------------------------
// finalize: out[i,o] /= max(cnt[i], 1)
// ---------------------------------------------------------------------------
__global__ __launch_bounds__(256) void cc_final(float* __restrict__ out,
                                                const float* __restrict__ cnt) {
    int i = blockIdx.x * 256 + threadIdx.x;
    if (i < N_NODES * COUT) out[i] /= fmaxf(cnt[i >> 5], 1.0f);  // COUT == 32
}

// ---------------------------------------------------------------------------
extern "C" void kernel_launch(void* const* d_in, const int* in_sizes, int n_in,
                              void* d_out, int out_size, void* d_ws, size_t ws_size,
                              hipStream_t stream) {
    const float* pos  = (const float*)d_in[0];   // (N,3)  f32
    const float* feat = (const float*)d_in[1];   // (N,16) f32
    const float* filt = (const float*)d_in[2];   // (4,4,4,16,32) f32
    const int*   row  = (const int*)d_in[3];     // (E,)   i32
    const int*   col  = (const int*)d_in[4];     // (E,)   i32
    float*       out  = (float*)d_out;           // (N,32) f32
    float*       cnt  = (float*)d_ws;            // N floats of scratch

    int initN = N_NODES * COUT + N_NODES;
    cc_init<<<(initN + 255) / 256, 256, 0, stream>>>(out, cnt);
    cc_edges<<<MAIN_BLOCKS, 256, 0, stream>>>(pos, feat, filt, row, col, out, cnt);
    cc_final<<<(N_NODES * COUT + 255) / 256, 256, 0, stream>>>(out, cnt);
}
